// Mamba3SSMLayer_78804059947097
// MI455X (gfx1250) — compile-verified, matrix operand order checked by reading, not run on hardware
//
#include <hip/hip_runtime.h>

#define D_MODEL 1024
#define D_STATE 32
#define D_INNER 2048
#define BATCH   2
#define SEQ     2048
#define M_TOT   (BATCH * SEQ)     // 4096
#define NPROJ   (2 * D_STATE + 1) // 65

typedef __attribute__((ext_vector_type(16))) __bf16 v16bf;
typedef __attribute__((ext_vector_type(8)))  __bf16 v8bf;
typedef __attribute__((ext_vector_type(8)))  float  v8f;

// ---------------- fp32 -> bf16 convert ----------------
__global__ void f32_to_bf16_kernel(const float* __restrict__ in,
                                   __bf16* __restrict__ out, int n) {
  int i = blockIdx.x * blockDim.x + threadIdx.x;
  if (i < n) out[i] = (__bf16)in[i];
}

// ---------------- WMMA bf16 GEMM: C = A(MxK) * W(NxK)^T ----------------
// Fragment layout per CDNA5 ISA 16-bit A 16x32:
//   lane l: M/N = l&15; kg = (l<16)?0:8; vec elems 0..7 <-> K=kg..kg+7,
//   elems 8..15 <-> K=kg+16..kg+23.
__device__ inline v16bf load_frag_bf16(const __bf16* __restrict__ p) {
  v8bf lo = *(const v8bf*)(p);        // K = kg .. kg+7
  v8bf hi = *(const v8bf*)(p + 16);   // K = kg+16 .. kg+23
  v16bf r;
#pragma unroll
  for (int i = 0; i < 8; ++i) { r[i] = lo[i]; r[i + 8] = hi[i]; }
  return r;
}

// Output split at column `split`:
//   n <  split -> C0[m*split + n]            (width split)
//   n >= split -> C1[m*(N-split) + n-split]  (width N-split)
// Each wave computes a 16x64 strip (4 WMMA tiles sharing one A fragment).
// k-loop is ping-pong double-buffered (unroll-by-2, disjoint register sets)
// so prefetch overlaps WMMA without cross-iteration register copies.
// Requires K % 64 == 0 (holds: K = 1024 or 2048).
__global__ __launch_bounds__(256)
void gemm_bf16_wmma_kernel(const __bf16* __restrict__ A,
                           const __bf16* __restrict__ W,
                           float* __restrict__ C0,
                           float* __restrict__ C1,
                           int M, int N, int K, int split) {
  const int lane  = threadIdx.x & 31;
  const int wave  = threadIdx.x >> 5;
  const int mbase = blockIdx.y * 16;
  const int nbase = (blockIdx.x * 8 + wave) * 64;
  const int half  = lane >> 4;     // 0 or 1
  const int mr    = lane & 15;
  const int kg    = half * 8;

  const __bf16* aptr = A + (size_t)(mbase + mr) * K + kg;
  const __bf16* wptr[4];
#pragma unroll
  for (int t = 0; t < 4; ++t)
    wptr[t] = W + (size_t)(nbase + t * 16 + mr) * K + kg;

  v8f acc[4] = {v8f{}, v8f{}, v8f{}, v8f{}};

  // set0 <- k = 0
  v16bf af0 = load_frag_bf16(aptr);
  v16bf b0[4];
#pragma unroll
  for (int t = 0; t < 4; ++t) b0[t] = load_frag_bf16(wptr[t]);

  v16bf af1;
  v16bf b1[4];

  int k0 = 0;
  for (; k0 < K - 64; k0 += 64) {
    // prefetch set1 <- k0+32 while consuming set0
    af1 = load_frag_bf16(aptr + k0 + 32);
#pragma unroll
    for (int t = 0; t < 4; ++t) b1[t] = load_frag_bf16(wptr[t] + k0 + 32);
#pragma unroll
    for (int t = 0; t < 4; ++t)
      acc[t] = __builtin_amdgcn_wmma_f32_16x16x32_bf16(
          false, af0, false, b0[t], (short)0, acc[t], false, false);

    // prefetch set0 <- k0+64 while consuming set1
    af0 = load_frag_bf16(aptr + k0 + 64);
#pragma unroll
    for (int t = 0; t < 4; ++t) b0[t] = load_frag_bf16(wptr[t] + k0 + 64);
#pragma unroll
    for (int t = 0; t < 4; ++t)
      acc[t] = __builtin_amdgcn_wmma_f32_16x16x32_bf16(
          false, af1, false, b1[t], (short)0, acc[t], false, false);
  }

  // tail: k0 == K-64; set0 holds slice K-64, load slice K-32 into set1
  af1 = load_frag_bf16(aptr + k0 + 32);
#pragma unroll
  for (int t = 0; t < 4; ++t) b1[t] = load_frag_bf16(wptr[t] + k0 + 32);
#pragma unroll
  for (int t = 0; t < 4; ++t)
    acc[t] = __builtin_amdgcn_wmma_f32_16x16x32_bf16(
        false, af0, false, b0[t], (short)0, acc[t], false, false);
#pragma unroll
  for (int t = 0; t < 4; ++t)
    acc[t] = __builtin_amdgcn_wmma_f32_16x16x32_bf16(
        false, af1, false, b1[t], (short)0, acc[t], false, false);

  // C/D layout: VGPR r -> M = r + 8*half, N = lane&15
#pragma unroll
  for (int t = 0; t < 4; ++t) {
    int ncol = nbase + t * 16 + mr;
#pragma unroll
    for (int r = 0; r < 8; ++r) {
      int mrow = mbase + r + (half << 3);
      float v = acc[t][r];
      if (ncol < split) C0[(size_t)mrow * split + ncol] = v;
      else              C1[(size_t)mrow * (N - split) + (ncol - split)] = v;
    }
  }
}

// ---------------- depthwise conv(3) + SiLU ----------------
__global__ void conv_silu_kernel(const float* __restrict__ x_in,
                                 const float* __restrict__ conv_w,
                                 const float* __restrict__ conv_b,
                                 float* __restrict__ x_conv) {
  int d = blockIdx.x * blockDim.x + threadIdx.x;  // 0..D_INNER-1
  int m = blockIdx.y;                             // 0..M_TOT-1
  int t = m % SEQ;
  float w0 = conv_w[d * 3 + 0], w1 = conv_w[d * 3 + 1], w2 = conv_w[d * 3 + 2];
  float acc = conv_b[d];
  if (t > 0)       acc += w0 * x_in[(size_t)(m - 1) * D_INNER + d];
  acc              += w1 * x_in[(size_t)m * D_INNER + d];
  if (t < SEQ - 1) acc += w2 * x_in[(size_t)(m + 1) * D_INNER + d];
  x_conv[(size_t)m * D_INNER + d] = acc / (1.f + __expf(-acc));  // silu
}

// ---------------- x-projection: proj = x_conv @ W_xproj^T (N = 65) ----------------
// 8 waves round-robin over the 65 outputs; lane-strided k + shfl reduction.
__global__ __launch_bounds__(256)
void xproj_kernel(const float* __restrict__ x_conv,
                  const float* __restrict__ W_xproj,
                  float* __restrict__ proj) {
  __shared__ float row[D_INNER];
  int m = blockIdx.x;
  for (int k = threadIdx.x; k < D_INNER; k += 256)
    row[k] = x_conv[(size_t)m * D_INNER + k];
  __syncthreads();
  int lane = threadIdx.x & 31;
  int wave = threadIdx.x >> 5;
  for (int n = wave; n < NPROJ; n += 8) {
    const float* w = W_xproj + (size_t)n * D_INNER;
    float acc = 0.f;
    for (int k = lane; k < D_INNER; k += 32) acc += row[k] * w[k];
#pragma unroll
    for (int off = 16; off > 0; off >>= 1) acc += __shfl_xor(acc, off, 32);
    if (lane == 0) proj[(size_t)m * NPROJ + n] = acc;
  }
}

// ---------------- selective scan: one wave32 per (b,d), lane = state ----------------
// Step t+1's operands are prefetched while step t's recurrence computes,
// hiding global-load latency under the serial critical path.
__global__ __launch_bounds__(256)
void scan_kernel(const float* __restrict__ proj,
                 const float* __restrict__ x_conv,
                 const float* __restrict__ zbuf,
                 const float* __restrict__ A_log,
                 const float* __restrict__ D_skip,
                 const float* __restrict__ W_dt,
                 const float* __restrict__ b_dt,
                 const float* __restrict__ alpha,
                 __bf16* __restrict__ ybf) {
  const int lane = threadIdx.x & 31;
  const int wg   = blockIdx.x * (blockDim.x >> 5) + (threadIdx.x >> 5);
  const int b = wg / D_INNER;
  const int d = wg % D_INNER;

  const float a     = 1.f / (1.f + __expf(-alpha[0]));
  const float Areal = -__expf(A_log[(size_t)d * D_STATE + lane]);
  const float wdt = W_dt[d];
  const float bdt = b_dt[d];
  const float Dd  = D_skip[d];

  float h = 0.f;
  const size_t mbase = (size_t)b * SEQ;

  // prologue loads for t = 0
  float dtraw = proj[mbase * NPROJ];
  float Bn    = proj[mbase * NPROJ + 1 + lane];
  float Cn    = proj[mbase * NPROJ + 1 + D_STATE + lane];
  float u     = x_conv[mbase * D_INNER + d];
  float zv    = zbuf[mbase * D_INNER + d];

  for (int t = 0; t < SEQ; ++t) {
    // prefetch step t+1 operands (clamped so loads stay in-bounds)
    const size_t mn = mbase + ((t + 1 < SEQ) ? t + 1 : t);
    float dtraw_n = proj[mn * NPROJ];
    float Bn_n    = proj[mn * NPROJ + 1 + lane];
    float Cn_n    = proj[mn * NPROJ + 1 + D_STATE + lane];
    float u_n     = x_conv[mn * D_INNER + d];
    float zv_n    = zbuf[mn * D_INNER + d];

    // recurrence for step t
    float xdt = dtraw * wdt + bdt;
    float dtv = (xdt > 20.f) ? xdt : log1pf(__expf(xdt));  // softplus
    float dtA = Areal * dtv;
    float Ad  = a * __expf(dtA) + (1.f - a) / (1.f - 0.5f * dtA + 1e-6f);
    h = Ad * h + (dtv * u) * Bn;

    float y = Cn * h;
#pragma unroll
    for (int off = 16; off > 0; off >>= 1) y += __shfl_xor(y, off, 32);

    if (lane == 0) {
      float g = zv / (1.f + __expf(-zv));  // silu(z)
      ybf[(mbase + t) * D_INNER + d] = (__bf16)((y + Dd * u) * g);
    }

    dtraw = dtraw_n; Bn = Bn_n; Cn = Cn_n; u = u_n; zv = zv_n;
  }
}

// ---------------- host launcher ----------------
extern "C" void kernel_launch(void* const* d_in, const int* in_sizes, int n_in,
                              void* d_out, int out_size, void* d_ws, size_t ws_size,
                              hipStream_t stream) {
  (void)in_sizes; (void)n_in; (void)out_size; (void)ws_size;
  const float* x       = (const float*)d_in[0];
  const float* W_in    = (const float*)d_in[1];
  const float* conv_w  = (const float*)d_in[2];
  const float* conv_b  = (const float*)d_in[3];
  const float* A_log   = (const float*)d_in[4];
  const float* D_skip  = (const float*)d_in[5];
  const float* W_xproj = (const float*)d_in[6];
  const float* W_dt    = (const float*)d_in[7];
  const float* b_dt    = (const float*)d_in[8];
  const float* alpha   = (const float*)d_in[9];
  const float* W_out   = (const float*)d_in[10];
  float* out = (float*)d_out;

  char* ws = (char*)d_ws;
  size_t off = 0;
  auto take = [&](size_t bytes) -> char* {
    char* p = ws + off;
    off += (bytes + 255) & ~(size_t)255;
    return p;
  };
  __bf16* xb     = (__bf16*)take((size_t)M_TOT * D_MODEL * 2);
  __bf16* Winb   = (__bf16*)take((size_t)2 * D_INNER * D_MODEL * 2);
  __bf16* Woutb  = (__bf16*)take((size_t)D_MODEL * D_INNER * 2);
  float*  x_in   = (float*) take((size_t)M_TOT * D_INNER * 4);
  float*  zbuf   = (float*) take((size_t)M_TOT * D_INNER * 4);
  float*  x_conv = (float*) take((size_t)M_TOT * D_INNER * 4);
  float*  proj   = (float*) take((size_t)M_TOT * NPROJ * 4);
  __bf16* ybf    = (__bf16*)take((size_t)M_TOT * D_INNER * 2);

  // 1) bf16 copies of x, W_in, W_out
  {
    int n1 = M_TOT * D_MODEL;
    f32_to_bf16_kernel<<<(n1 + 255) / 256, 256, 0, stream>>>(x, xb, n1);
    int n2 = 2 * D_INNER * D_MODEL;
    f32_to_bf16_kernel<<<(n2 + 255) / 256, 256, 0, stream>>>(W_in, Winb, n2);
    int n3 = D_MODEL * D_INNER;
    f32_to_bf16_kernel<<<(n3 + 255) / 256, 256, 0, stream>>>(W_out, Woutb, n3);
  }

  // 2) xz = x @ W_in^T  (M=4096, N=4096, K=1024), split into x_in | z
  {
    dim3 grid((2 * D_INNER) / 512, M_TOT / 16);  // (8, 256)
    gemm_bf16_wmma_kernel<<<grid, 256, 0, stream>>>(
        xb, Winb, x_in, zbuf, M_TOT, 2 * D_INNER, D_MODEL, D_INNER);
  }

  // 3) depthwise conv + silu
  {
    dim3 grid(D_INNER / 256, M_TOT);
    conv_silu_kernel<<<grid, 256, 0, stream>>>(x_in, conv_w, conv_b, x_conv);
  }

  // 4) proj = x_conv @ W_xproj^T (N=65)
  xproj_kernel<<<M_TOT, 256, 0, stream>>>(x_conv, W_xproj, proj);

  // 5) selective scan + gating, writes bf16 y
  scan_kernel<<<(BATCH * D_INNER) / 8, 256, 0, stream>>>(
      proj, x_conv, zbuf, A_log, D_skip, W_dt, b_dt, alpha, ybf);

  // 6) out = y @ W_out^T  (M=4096, N=1024, K=2048)
  {
    dim3 grid(D_MODEL / 512, M_TOT / 16);  // (2, 256)
    gemm_bf16_wmma_kernel<<<grid, 256, 0, stream>>>(
        ybf, Woutb, out, out, M_TOT, D_MODEL, D_INNER, D_MODEL);
  }
}
